// GAT_8753143349916
// MI455X (gfx1250) — matrix operand order repaired and checked by reading, not compiled, not run on hardware
//
#include <hip/hip_runtime.h>
#include <hip/hip_bf16.h>

typedef __attribute__((ext_vector_type(16))) _Float16 v16h;
typedef __attribute__((ext_vector_type(8)))  float    v8f;

#define NEG_SLOPE 0.2f

// ---------------------------------------------------------------------------
// Fill a buffer with a 32-bit pattern (used for zeroing and -inf init).
// ---------------------------------------------------------------------------
__global__ void fill_u32_kernel(unsigned int* __restrict__ p, unsigned int val, int n) {
    int i = blockIdx.x * blockDim.x + threadIdx.x;
    if (i < n) p[i] = val;
}

// ---------------------------------------------------------------------------
// WMMA GEMM: C[Nr,64] = A[Nr,K] @ B[K,64]   (f32 in/out, f16 WMMA inputs)
// Block = 256 threads = 8 waves. Each wave: one 16-row tile x four 16-col
// tiles. B is staged in LDS pre-swizzled into WMMA B-fragment order so each
// lane's 16 f16 operand values are contiguous (32B -> 2x ds_load_b128).
// A fragments are unconditional float4 loads (row index clamped; surplus
// rows feed only output rows that are never stored).
// Fragment layouts used here (wave32, contiguous K order in A and B):
//   A 16x32 f16: lanes 0-15 row M=lane, elems = K 0..15;
//                lanes 16-31 row M=lane-16, elems = K 16..31
//   B 32x16 f16: lanes 0-15 col N=lane, elems = K 0..15;
//                lanes 16-31 col N=lane-16, elems = K 16..31
//   C 16x16 f32: VGPR v, lanes 0-15 -> (M=v, N=lane); lanes 16-31 -> (M=v+8)
// ---------------------------------------------------------------------------
template <int K>
__global__ __launch_bounds__(256)
void gemm_nodes_kernel(const float* __restrict__ A, const float* __restrict__ B,
                       float* __restrict__ C, int Nr) {
    // Slot layout: [(kk*4 + nt)][lane][i] ; 16 contiguous f16 per (frag,lane).
    __shared__ __attribute__((aligned(32))) _Float16 Bs[(K / 32) * 4 * 32 * 16];
    for (int idx = threadIdx.x; idx < K * 64; idx += 256) {
        int k    = idx >> 6;          // 0..K-1
        int n    = idx & 63;          // 0..63
        int kk   = k >> 5;            // 32-wide K block
        int kr   = k & 31;
        int hf   = kr >> 4;           // which lane half holds this K
        int i    = kr & 15;           // element index within fragment
        int nt   = n >> 4;            // 16-wide column tile
        int lane = hf * 16 + (n & 15);
        Bs[((((kk << 2) + nt) << 5) + lane) * 16 + i] = (_Float16)B[idx];
    }
    __syncthreads();

    const int wave = threadIdx.x >> 5;
    const int lane = threadIdx.x & 31;
    const int half = lane >> 4;   // 0 or 1
    const int m    = lane & 15;
    const int r0   = blockIdx.x * 128 + wave * 16;
    if (r0 >= Nr) return;

    v8f acc0 = {}, acc1 = {}, acc2 = {}, acc3 = {};

    int row = r0 + m;
    if (row >= Nr) row = Nr - 1;          // clamp: surplus rows never stored
    const float* arow = A + (size_t)row * K;

#pragma unroll
    for (int kk = 0; kk < K / 32; ++kk) {
        // ---- A fragment: 4 unconditional float4 (global_load_b128) ----
        const float4* ap = (const float4*)(arow + kk * 32);
        float4 q0 = ap[half * 4 + 0];     // K = half*16 + {0..3}
        float4 q1 = ap[half * 4 + 1];     // K = half*16 + {4..7}
        float4 q2 = ap[half * 4 + 2];     // K = half*16 + {8..11}
        float4 q3 = ap[half * 4 + 3];     // K = half*16 + {12..15}
        v16h a;
        a[0]  = (_Float16)q0.x; a[1]  = (_Float16)q0.y;
        a[2]  = (_Float16)q0.z; a[3]  = (_Float16)q0.w;
        a[4]  = (_Float16)q1.x; a[5]  = (_Float16)q1.y;
        a[6]  = (_Float16)q1.z; a[7]  = (_Float16)q1.w;
        a[8]  = (_Float16)q2.x; a[9]  = (_Float16)q2.y;
        a[10] = (_Float16)q2.z; a[11] = (_Float16)q2.w;
        a[12] = (_Float16)q3.x; a[13] = (_Float16)q3.y;
        a[14] = (_Float16)q3.z; a[15] = (_Float16)q3.w;

        // ---- B fragments: contiguous 32B per lane, 4 column tiles ----
        const _Float16* bb = Bs + (kk << 11) + (lane << 4);
        v16h b0 = *(const v16h*)(bb + (0 << 9));
        v16h b1 = *(const v16h*)(bb + (1 << 9));
        v16h b2 = *(const v16h*)(bb + (2 << 9));
        v16h b3 = *(const v16h*)(bb + (3 << 9));

        acc0 = __builtin_amdgcn_wmma_f32_16x16x32_f16(false, a, false, b0, (short)0, acc0, false, false);
        acc1 = __builtin_amdgcn_wmma_f32_16x16x32_f16(false, a, false, b1, (short)0, acc1, false, false);
        acc2 = __builtin_amdgcn_wmma_f32_16x16x32_f16(false, a, false, b2, (short)0, acc2, false, false);
        acc3 = __builtin_amdgcn_wmma_f32_16x16x32_f16(false, a, false, b3, (short)0, acc3, false, false);
    }

    // ---- store C tile ----
#pragma unroll
    for (int v = 0; v < 8; ++v) {
        int rr = r0 + half * 8 + v;
        if (rr < Nr) {
            float* crow = C + (size_t)rr * 64 + m;
            crow[0]  = acc0[v];
            crow[16] = acc1[v];
            crow[32] = acc2[v];
            crow[48] = acc3[v];
        }
    }
}

// ---------------------------------------------------------------------------
// Per (node, head): a_src[n,h] = dot(H[n,h,:], att_src[h,:]) (likewise dst)
// ---------------------------------------------------------------------------
__global__ void attn_dot_kernel(const float* __restrict__ H,
                                const float* __restrict__ att_s,
                                const float* __restrict__ att_d,
                                float* __restrict__ as_, float* __restrict__ ad_,
                                int N, int heads, int C) {
    int idx = blockIdx.x * blockDim.x + threadIdx.x;
    if (idx >= N * heads) return;
    int n = idx / heads, hd = idx - n * heads;
    const float4* hp = (const float4*)(H + (size_t)n * heads * C + hd * C);
    const float4* sp = (const float4*)(att_s + hd * C);
    const float4* dp = (const float4*)(att_d + hd * C);
    float s = 0.f, d = 0.f;
    for (int c = 0; c < C / 4; ++c) {
        float4 hv = hp[c], sv = sp[c], dv = dp[c];
        s += hv.x * sv.x + hv.y * sv.y + hv.z * sv.z + hv.w * sv.w;
        d += hv.x * dv.x + hv.y * dv.y + hv.z * dv.z + hv.w * dv.w;
    }
    as_[idx] = s;
    ad_[idx] = d;
}

__device__ __forceinline__ void edge_nodes(const int* __restrict__ ei, int E,
                                           int e, int& s, int& d) {
    if (e < E) { s = ei[e]; d = ei[E + e]; }
    else       { s = d = e - E; }   // self loops appended
}

// ---------------------------------------------------------------------------
// Edge pass 1: e = leaky_relu(a_src[src] + a_dst[dst]); segment-max into m[dst]
// Float atomic max via signed/unsigned int-bit trick (init bits 0xFF800000).
// ---------------------------------------------------------------------------
__global__ void edge_max_kernel(const int* __restrict__ ei, int E, int N, int heads,
                                const float* __restrict__ as_, const float* __restrict__ ad_,
                                float* __restrict__ ebuf, float* __restrict__ mbuf) {
    int idx = blockIdx.x * blockDim.x + threadIdx.x;
    int total = (E + N) * heads;
    if (idx >= total) return;
    int e = idx / heads, hd = idx - e * heads;
    int s, d;
    edge_nodes(ei, E, e, s, d);
    float v = as_[s * heads + hd] + ad_[d * heads + hd];
    v = (v > 0.f) ? v : NEG_SLOPE * v;
    ebuf[idx] = v;
    int* addr = (int*)(mbuf + (size_t)d * heads + hd);
    int vi = __float_as_int(v);
    if (vi >= 0) atomicMax(addr, vi);
    else         atomicMin((unsigned int*)addr, (unsigned int)vi);
}

// ---------------------------------------------------------------------------
// Edge pass 2: e <- exp(e - m[dst]); denom[dst] += e
// ---------------------------------------------------------------------------
__global__ void edge_exp_kernel(const int* __restrict__ ei, int E, int N, int heads,
                                float* __restrict__ ebuf, const float* __restrict__ mbuf,
                                float* __restrict__ den) {
    int idx = blockIdx.x * blockDim.x + threadIdx.x;
    int total = (E + N) * heads;
    if (idx >= total) return;
    int e = idx / heads, hd = idx - e * heads;
    int s, d;
    edge_nodes(ei, E, e, s, d);
    float v = __expf(ebuf[idx] - mbuf[(size_t)d * heads + hd]);
    ebuf[idx] = v;
    atomicAdd(den + (size_t)d * heads + hd, v);
}

// ---------------------------------------------------------------------------
// Edge pass 3: out[dst, c] += feat[src, c] * (e / denom[dst]); thread=(edge,c)
// 64 channels per edge -> coalesced feature reads and L2-resident atomics.
// ---------------------------------------------------------------------------
__global__ void edge_scatter_kernel(const int* __restrict__ ei, int E, int N, int heads,
                                    const float* __restrict__ feat,
                                    const float* __restrict__ ebuf,
                                    const float* __restrict__ den,
                                    float* __restrict__ out) {
    int idx = blockIdx.x * blockDim.x + threadIdx.x;
    int total = (E + N) * 64;
    if (idx >= total) return;
    int c = idx & 63;
    int e = idx >> 6;
    int hd = (heads == 2) ? (c >> 5) : 0;
    int s, d;
    edge_nodes(ei, E, e, s, d);
    float alpha = ebuf[(size_t)e * heads + hd] / den[(size_t)d * heads + hd];
    atomicAdd(out + (size_t)d * 64 + c, feat[(size_t)s * 64 + c] * alpha);
}

// ---------------------------------------------------------------------------
// h <- elu(h + bias)   (in place, 64-wide rows)
// ---------------------------------------------------------------------------
__global__ void bias_elu_kernel(float* __restrict__ h, const float* __restrict__ b, int total) {
    int idx = blockIdx.x * blockDim.x + threadIdx.x;
    if (idx >= total) return;
    float v = h[idx] + b[idx & 63];
    h[idx] = (v > 0.f) ? v : (__expf(v) - 1.f);
}

// ---------------------------------------------------------------------------
// out <- agg + bias
// ---------------------------------------------------------------------------
__global__ void bias_out_kernel(const float* __restrict__ agg, const float* __restrict__ b,
                                float* __restrict__ out, int total) {
    int idx = blockIdx.x * blockDim.x + threadIdx.x;
    if (idx >= total) return;
    out[idx] = agg[idx] + b[idx & 63];
}

// ---------------------------------------------------------------------------
// Host-side orchestration (graph-capture safe: launches only).
// ---------------------------------------------------------------------------
extern "C" void kernel_launch(void* const* d_in, const int* in_sizes, int n_in,
                              void* d_out, int out_size, void* d_ws, size_t ws_size,
                              hipStream_t stream) {
    const float* x        = (const float*)d_in[0];   // [N,128]
    const int*   ei       = (const int*)  d_in[1];   // [2,E]
    const float* W1       = (const float*)d_in[2];   // [128,64]
    const float* att_src1 = (const float*)d_in[3];   // [2,32]
    const float* att_dst1 = (const float*)d_in[4];   // [2,32]
    const float* b1       = (const float*)d_in[5];   // [64]
    const float* W2       = (const float*)d_in[6];   // [64,64]
    const float* att_src2 = (const float*)d_in[7];   // [1,64]
    const float* att_dst2 = (const float*)d_in[8];   // [1,64]
    const float* b2       = (const float*)d_in[9];   // [64]

    const int N  = in_sizes[0] / 128;
    const int E  = in_sizes[1] / 2;
    const int ET = E + N;                 // with self loops

    // ---- workspace carve (floats) ----
    float* ws   = (float*)d_ws;
    float* h1   = ws;                     // [N,64]  layer-1 features; reused as g2
    float* agg  = h1  + (size_t)N * 64;   // [N,64]  agg1 -> h2 -> agg2
    float* ebuf = agg + (size_t)N * 64;   // [ET,2]  per-edge e values (layer2 uses first ET)
    float* as_  = ebuf + (size_t)ET * 2;  // [N,2]
    float* ad_  = as_ + (size_t)N * 2;    // [N,2]
    float* mb   = ad_ + (size_t)N * 2;    // [N,2]
    float* den  = mb  + (size_t)N * 2;    // [N,2]

    const int T = 256;
    const unsigned NEG_INF_BITS = 0xFF800000u;
    #define GRID(n) dim3(((n) + T - 1) / T)

    // ================= Layer 1 (heads=2, C=32, concat) =================
    fill_u32_kernel<<<GRID(N * 64), T, 0, stream>>>((unsigned*)agg, 0u, N * 64);
    fill_u32_kernel<<<GRID(N * 2),  T, 0, stream>>>((unsigned*)mb,  NEG_INF_BITS, N * 2);
    fill_u32_kernel<<<GRID(N * 2),  T, 0, stream>>>((unsigned*)den, 0u, N * 2);

    gemm_nodes_kernel<128><<<dim3((N + 127) / 128), T, 0, stream>>>(x, W1, h1, N);
    attn_dot_kernel<<<GRID(N * 2), T, 0, stream>>>(h1, att_src1, att_dst1, as_, ad_, N, 2, 32);

    edge_max_kernel    <<<GRID(ET * 2),  T, 0, stream>>>(ei, E, N, 2, as_, ad_, ebuf, mb);
    edge_exp_kernel    <<<GRID(ET * 2),  T, 0, stream>>>(ei, E, N, 2, ebuf, mb, den);
    edge_scatter_kernel<<<GRID(ET * 64), T, 0, stream>>>(ei, E, N, 2, h1, ebuf, den, agg);

    bias_elu_kernel<<<GRID(N * 64), T, 0, stream>>>(agg, b1, N * 64);   // agg is now h2

    // ================= Layer 2 (heads=1, C=64, mean==identity) =========
    gemm_nodes_kernel<64><<<dim3((N + 127) / 128), T, 0, stream>>>(agg, W2, h1, N); // g2 in h1
    attn_dot_kernel<<<GRID(N), T, 0, stream>>>(h1, att_src2, att_dst2, as_, ad_, N, 1, 64);

    fill_u32_kernel<<<GRID(N * 64), T, 0, stream>>>((unsigned*)agg, 0u, N * 64);    // agg2
    fill_u32_kernel<<<GRID(N), T, 0, stream>>>((unsigned*)mb,  NEG_INF_BITS, N);
    fill_u32_kernel<<<GRID(N), T, 0, stream>>>((unsigned*)den, 0u, N);

    edge_max_kernel    <<<GRID(ET),      T, 0, stream>>>(ei, E, N, 1, as_, ad_, ebuf, mb);
    edge_exp_kernel    <<<GRID(ET),      T, 0, stream>>>(ei, E, N, 1, ebuf, mb, den);
    edge_scatter_kernel<<<GRID(ET * 64), T, 0, stream>>>(ei, E, N, 1, h1, ebuf, den, agg);

    bias_out_kernel<<<GRID(N * 64), T, 0, stream>>>(agg, b2, (float*)d_out, N * 64);
    #undef GRID
}
